// Informer_SpliceAI_18665927868721
// MI455X (gfx1250) — matrix operand order repaired and checked by reading, not compiled, not run on hardware
//
#include <hip/hip_runtime.h>
#include <hip/hip_bf16.h>

#define B_    4
#define L_    2048
#define DM_   512
#define H_    8
#define DK_   64
#define DFF_  2048
#define E_    3
#define U_    40
#define R_    (B_*L_)          // 8192 token rows
#define SCALE_ 0.125f          // 1/sqrt(64)
#define EPS_   1e-5f

typedef __attribute__((ext_vector_type(16))) __bf16 v16bf;
typedef __attribute__((ext_vector_type(8)))  float  v8f;

union ABFrag { uint4 u4[2]; v16bf v; };

// float -> bf16 (round-to-nearest-even), as raw ushort
static __device__ __forceinline__ unsigned short f2bf(float f) {
    unsigned int u = __float_as_uint(f);
    u += 0x7FFFu + ((u >> 16) & 1u);
    return (unsigned short)(u >> 16);
}

// -------------------------------------------------------------------------
// WMMA GEMM: Y[M,N] = X[M,K] @ W[K,N] + bias, optional ReLU.
// Block = 256 threads (8 waves). Block tile 128x128, wave tile 32x64
// (2x4 WMMA accumulators -> 8 v_wmma per k-step per wave).
// Software pipeline: global loads for tile s+1 are issued into a register
// buffer BEFORE the WMMAs of tile s, and only converted/stored to the LDS
// ping-pong buffer AFTER them, so s_wait_loadcnt lands after the matrix ops
// and global latency hides under compute. One barrier per k-step.
// -------------------------------------------------------------------------
__global__ __launch_bounds__(256)
void gemm_wmma_bf16(const float* __restrict__ X, const float* __restrict__ W,
                    const float* __restrict__ bias, float* __restrict__ Y,
                    int M, int N, int K, int relu)
{
    __shared__ __align__(16) unsigned short lA[2][128 * 32];  // [m][k] 2x8KB
    __shared__ __align__(16) unsigned short lB[2][128 * 32];  // [n][k] 2x8KB (transposed)

    const int tid  = threadIdx.x;
    const int bm   = blockIdx.x;   // M / 128
    const int bn   = blockIdx.y;   // N / 128
    const int wave = tid >> 5;
    const int lane = tid & 31;
    const int wm   = (wave & 3) * 32;   // 0,32,64,96
    const int wn   = (wave >> 2) * 64;  // 0,64
    const int lm   = lane & 15;
    const int half = lane >> 4;

    v8f acc[2][4];
    #pragma unroll
    for (int i = 0; i < 2; ++i)
        #pragma unroll
        for (int j = 0; j < 4; ++j)
            #pragma unroll
            for (int e = 0; e < 8; ++e) acc[i][j][e] = 0.0f;

    // ---- staging: load phase (global -> regs, no wait consumed) ----
    auto loadTiles = [&](int k0, float4* ra, float4* rb) {
        const float* Xg = X + (size_t)(bm * 128) * K + k0;
        #pragma unroll
        for (int t = 0; t < 4; ++t) {
            int f   = tid + t * 256;        // 1024 float4 slots (128x32 f32)
            int row = f >> 3;               // 8 float4 per 32-wide row
            int c4  = f & 7;
            ra[t] = *reinterpret_cast<const float4*>(Xg + (size_t)row * K + c4 * 4);
        }
        const float* Wg = W + (size_t)k0 * N + bn * 128;
        #pragma unroll
        for (int t = 0; t < 4; ++t) {
            int f  = tid + t * 256;         // 1024 float4 slots (32x128 f32)
            int kk = f >> 5;                // 32 float4 per 128-wide row
            int nb = f & 31;
            rb[t] = *reinterpret_cast<const float4*>(Wg + (size_t)kk * N + nb * 4);
        }
    };
    // ---- staging: convert + store phase (regs -> bf16 LDS) ----
    auto storeTiles = [&](int buf, const float4* ra, const float4* rb) {
        #pragma unroll
        for (int t = 0; t < 4; ++t) {
            int f   = tid + t * 256;
            int row = f >> 3;
            int c4  = f & 7;
            unsigned short* d = &lA[buf][row * 32 + c4 * 4];
            d[0] = f2bf(ra[t].x); d[1] = f2bf(ra[t].y);
            d[2] = f2bf(ra[t].z); d[3] = f2bf(ra[t].w);
        }
        #pragma unroll
        for (int t = 0; t < 4; ++t) {
            int f  = tid + t * 256;
            int kk = f >> 5;
            int nb = f & 31;
            lB[buf][(nb * 4 + 0) * 32 + kk] = f2bf(rb[t].x);
            lB[buf][(nb * 4 + 1) * 32 + kk] = f2bf(rb[t].y);
            lB[buf][(nb * 4 + 2) * 32 + kk] = f2bf(rb[t].z);
            lB[buf][(nb * 4 + 3) * 32 + kk] = f2bf(rb[t].w);
        }
    };

    float4 ra[4], rb[4];
    loadTiles(0, ra, rb);
    storeTiles(0, ra, rb);
    __syncthreads();

    const int steps = K >> 5;
    for (int s = 0; s < steps; ++s) {
        const int  cur = s & 1, nxt = cur ^ 1;
        const bool pf  = (s + 1 < steps);

        // 1) issue next tile's global loads (clause; latency hides under WMMAs)
        if (pf) loadTiles((s + 1) * 32, ra, rb);

        // 2) fragment loads + matrix ops on current tile
        const uint4* pA = reinterpret_cast<const uint4*>(lA[cur]);
        const uint4* pB = reinterpret_cast<const uint4*>(lB[cur]);
        ABFrag a[2], b[4];
        #pragma unroll
        for (int i = 0; i < 2; ++i) {
            int m = wm + i * 16 + lm;
            // lane holds K = half*8..+7 (V0-3) and 16+half*8..+7 (V4-7)
            a[i].u4[0] = pA[m * 4 + half];
            a[i].u4[1] = pA[m * 4 + 2 + half];
        }
        #pragma unroll
        for (int j = 0; j < 4; ++j) {
            int n = wn + j * 16 + lm;
            // lane holds K = half*8..+7 (V0-3) and 16+half*8..+7 (V4-7)
            b[j].u4[0] = pB[n * 4 + half];
            b[j].u4[1] = pB[n * 4 + 2 + half];
        }
        #pragma unroll
        for (int i = 0; i < 2; ++i)
            #pragma unroll
            for (int j = 0; j < 4; ++j)
                acc[i][j] = __builtin_amdgcn_wmma_f32_16x16x32_bf16(
                    false, a[i].v, false, b[j].v, (short)0, acc[i][j], false, false);

        // 3) convert + store next tile into the other LDS buffer
        if (pf) storeTiles(nxt, ra, rb);
        __syncthreads();   // single barrier per k-step (ping-pong)
    }

    // ---- epilogue: bias (+ReLU), C layout: VGPR r -> M = half*8 + r ----
    #pragma unroll
    for (int i = 0; i < 2; ++i)
        #pragma unroll
        for (int j = 0; j < 4; ++j) {
            int nc = bn * 128 + wn + j * 16 + lm;
            float bv = bias[nc];
            #pragma unroll
            for (int r = 0; r < 8; ++r) {
                int mr = bm * 128 + wm + i * 16 + half * 8 + r;
                float val = acc[i][j][r] + bv;
                if (relu) val = fmaxf(val, 0.0f);
                Y[(size_t)mr * N + nc] = val;
            }
        }
}

// -------------------------------------------------------------------------
// x = DNAs @ W_pre + b_pre  (K=4, plain VALU)
// -------------------------------------------------------------------------
__global__ __launch_bounds__(256)
void embed_kernel(const float* __restrict__ DNAs, const float* __restrict__ Wp,
                  const float* __restrict__ bp, float* __restrict__ x)
{
    int t   = blockIdx.x * 256 + threadIdx.x;   // R_*DM_
    int row = t / DM_, n = t % DM_;
    const float* d = DNAs + (size_t)row * 4;
    float s = bp[n];
    #pragma unroll
    for (int c = 0; c < 4; ++c) s = fmaf(d[c], Wp[c * DM_ + n], s);
    x[t] = s;
}

// -------------------------------------------------------------------------
// M[b,h,l] = max_u qk - mean_u qk over sampled keys
// -------------------------------------------------------------------------
__global__ __launch_bounds__(256)
void measure_kernel(const float* __restrict__ q, const float* __restrict__ k,
                    const int* __restrict__ sidx, float* __restrict__ Mout)
{
    int t = blockIdx.x * 256 + threadIdx.x;     // B_*H_*L_
    int l = t % L_;
    int h = (t / L_) % H_;
    int b = t / (H_ * L_);
    const float* qr = q + (size_t)(b * L_ + l) * DM_ + h * DK_;
    float qreg[DK_];
    #pragma unroll
    for (int d = 0; d < DK_; ++d) qreg[d] = qr[d];
    float mx = -1e30f, sm = 0.0f;
    for (int u = 0; u < U_; ++u) {
        int kidx = sidx[l * U_ + u];
        const float* kr = k + (size_t)(b * L_ + kidx) * DM_ + h * DK_;
        float dot = 0.0f;
        #pragma unroll
        for (int d = 0; d < DK_; ++d) dot = fmaf(qreg[d], kr[d], dot);
        mx = fmaxf(mx, dot);
        sm += dot;
    }
    Mout[t] = mx - sm * (1.0f / (float)U_);
}

// -------------------------------------------------------------------------
// top-U selection per (b,h): iterative argmax over L=2048
// -------------------------------------------------------------------------
__global__ __launch_bounds__(256)
void topk_kernel(const float* __restrict__ Mbuf, int* __restrict__ top)
{
    __shared__ float buf[L_];
    __shared__ float rv[256];
    __shared__ int   ri[256];
    int bh  = blockIdx.x;                        // b*H+h
    int tid = threadIdx.x;
    const float* Mr = Mbuf + (size_t)bh * L_;
    for (int i = tid; i < L_; i += 256) buf[i] = Mr[i];
    __syncthreads();
    for (int u = 0; u < U_; ++u) {
        float bv = -1e30f; int bi = 0;
        for (int i = tid; i < L_; i += 256) {
            float xv = buf[i];
            if (xv > bv) { bv = xv; bi = i; }
        }
        rv[tid] = bv; ri[tid] = bi;
        __syncthreads();
        for (int s = 128; s > 0; s >>= 1) {
            if (tid < s && rv[tid + s] > rv[tid]) { rv[tid] = rv[tid + s]; ri[tid] = ri[tid + s]; }
            __syncthreads();
        }
        if (tid == 0) { top[bh * U_ + u] = ri[0]; buf[ri[0]] = -1e30f; }
        __syncthreads();
    }
}

// -------------------------------------------------------------------------
// vmean[b,h,d] = mean_l v[b,l,h,d]
// -------------------------------------------------------------------------
__global__ __launch_bounds__(256)
void vmean_kernel(const float* __restrict__ v, float* __restrict__ vmean)
{
    int t = blockIdx.x * 256 + threadIdx.x;     // B_*H_*DK_ = 2048
    int d = t % DK_;
    int h = (t / DK_) % H_;
    int b = t / (H_ * DK_);
    float s = 0.0f;
    for (int l = 0; l < L_; ++l) s += v[(size_t)(b * L_ + l) * DM_ + h * DK_ + d];
    vmean[t] = s * (1.0f / (float)L_);
}

// ctx[b,l,h,d] = vmean[b,h,d]  (ctx stored as flat (B,L,DM))
__global__ __launch_bounds__(256)
void ctxfill_kernel(const float* __restrict__ vmean, float* __restrict__ ctx)
{
    size_t t = (size_t)blockIdx.x * 256 + threadIdx.x;  // R_*DM_
    int dm = (int)(t % DM_);
    int b  = (int)(t / ((size_t)L_ * DM_));
    ctx[t] = vmean[(b * H_ + dm / DK_) * DK_ + (dm % DK_)];
}

// -------------------------------------------------------------------------
// For each selected query row: softmax(q.K^T * scale) @ V -> scatter into ctx
// -------------------------------------------------------------------------
__global__ __launch_bounds__(256)
void attn_update_kernel(const float* __restrict__ q, const float* __restrict__ k,
                        const float* __restrict__ v, const int* __restrict__ top,
                        float* __restrict__ ctx)
{
    __shared__ float sc[L_];
    __shared__ float qsh[DK_];
    __shared__ float red[256];
    __shared__ float part[256];
    int u = blockIdx.x % U_;
    int h = (blockIdx.x / U_) % H_;
    int b = blockIdx.x / (U_ * H_);
    int r = top[(b * H_ + h) * U_ + u];
    int tid = threadIdx.x;

    if (tid < DK_) qsh[tid] = q[(size_t)(b * L_ + r) * DM_ + h * DK_ + tid];
    __syncthreads();

    float lmx = -1e30f;
    for (int kk = tid; kk < L_; kk += 256) {
        const float* kr = k + (size_t)(b * L_ + kk) * DM_ + h * DK_;
        float dot = 0.0f;
        #pragma unroll
        for (int d = 0; d < DK_; ++d) dot = fmaf(qsh[d], kr[d], dot);
        dot *= SCALE_;
        sc[kk] = dot;
        lmx = fmaxf(lmx, dot);
    }
    red[tid] = lmx; __syncthreads();
    for (int s = 128; s > 0; s >>= 1) {
        if (tid < s) red[tid] = fmaxf(red[tid], red[tid + s]);
        __syncthreads();
    }
    float mx = red[0];
    __syncthreads();

    float lsum = 0.0f;
    for (int kk = tid; kk < L_; kk += 256) {
        float e = __expf(sc[kk] - mx);
        sc[kk] = e;
        lsum += e;
    }
    red[tid] = lsum; __syncthreads();
    for (int s = 128; s > 0; s >>= 1) {
        if (tid < s) red[tid] += red[tid + s];
        __syncthreads();
    }
    float ssum = red[0];

    int d = tid & 63, chunk = tid >> 6;         // 4 chunks of 512 keys
    float p = 0.0f;
    for (int kk = chunk * 512; kk < (chunk + 1) * 512; ++kk)
        p = fmaf(sc[kk], v[(size_t)(b * L_ + kk) * DM_ + h * DK_ + d], p);
    part[tid] = p; __syncthreads();
    if (tid < 64) {
        float tot = part[tid] + part[tid + 64] + part[tid + 128] + part[tid + 192];
        ctx[(size_t)(b * L_ + r) * DM_ + h * DK_ + tid] = tot / ssum;
    }
}

// -------------------------------------------------------------------------
// x = layer_norm(x + a) * g + b   (row = 512, block = 256, 2 elems/thread)
// -------------------------------------------------------------------------
__global__ __launch_bounds__(256)
void add_ln_kernel(float* __restrict__ x, const float* __restrict__ a,
                   const float* __restrict__ g, const float* __restrict__ bta)
{
    __shared__ float red[256];
    int r = blockIdx.x, tid = threadIdx.x;
    size_t base = (size_t)r * DM_;
    float s0 = x[base + tid]       + a[base + tid];
    float s1 = x[base + tid + 256] + a[base + tid + 256];
    red[tid] = s0 + s1; __syncthreads();
    for (int s = 128; s > 0; s >>= 1) { if (tid < s) red[tid] += red[tid + s]; __syncthreads(); }
    float mean = red[0] * (1.0f / (float)DM_);
    __syncthreads();
    float d0 = s0 - mean, d1 = s1 - mean;
    red[tid] = d0 * d0 + d1 * d1; __syncthreads();
    for (int s = 128; s > 0; s >>= 1) { if (tid < s) red[tid] += red[tid + s]; __syncthreads(); }
    float inv = rsqrtf(red[0] * (1.0f / (float)DM_) + EPS_);
    x[base + tid]       = d0 * inv * g[tid]       + bta[tid];
    x[base + tid + 256] = d1 * inv * g[tid + 256] + bta[tid + 256];
}

// -------------------------------------------------------------------------
// out = softmax(layer_norm(x) @ W_out + b_out), per row
// -------------------------------------------------------------------------
__global__ __launch_bounds__(256)
void final_kernel(const float* __restrict__ x, const float* __restrict__ g,
                  const float* __restrict__ bt, const float* __restrict__ Wout,
                  const float* __restrict__ bout, float* __restrict__ out)
{
    __shared__ float red[256];
    __shared__ float lg[3];
    int r = blockIdx.x, tid = threadIdx.x;
    size_t base = (size_t)r * DM_;
    float s0 = x[base + tid], s1 = x[base + tid + 256];
    red[tid] = s0 + s1; __syncthreads();
    for (int s = 128; s > 0; s >>= 1) { if (tid < s) red[tid] += red[tid + s]; __syncthreads(); }
    float mean = red[0] * (1.0f / (float)DM_);
    __syncthreads();
    float d0 = s0 - mean, d1 = s1 - mean;
    red[tid] = d0 * d0 + d1 * d1; __syncthreads();
    for (int s = 128; s > 0; s >>= 1) { if (tid < s) red[tid] += red[tid + s]; __syncthreads(); }
    float inv = rsqrtf(red[0] * (1.0f / (float)DM_) + EPS_);
    float n0 = d0 * inv * g[tid]       + bt[tid];
    float n1 = d1 * inv * g[tid + 256] + bt[tid + 256];
    float p[3];
    #pragma unroll
    for (int o = 0; o < 3; ++o)
        p[o] = n0 * Wout[tid * 3 + o] + n1 * Wout[(tid + 256) * 3 + o];
    for (int o = 0; o < 3; ++o) {
        __syncthreads();
        red[tid] = p[o]; __syncthreads();
        for (int s = 128; s > 0; s >>= 1) { if (tid < s) red[tid] += red[tid + s]; __syncthreads(); }
        if (tid == 0) lg[o] = red[0] + bout[o];
    }
    __syncthreads();
    if (tid == 0) {
        float m = fmaxf(lg[0], fmaxf(lg[1], lg[2]));
        float e0 = __expf(lg[0] - m), e1 = __expf(lg[1] - m), e2 = __expf(lg[2] - m);
        float s = e0 + e1 + e2;
        out[(size_t)r * 3 + 0] = e0 / s;
        out[(size_t)r * 3 + 1] = e1 / s;
        out[(size_t)r * 3 + 2] = e2 / s;
    }
}

// -------------------------------------------------------------------------
extern "C" void kernel_launch(void* const* d_in, const int* in_sizes, int n_in,
                              void* d_out, int out_size, void* d_ws, size_t ws_size,
                              hipStream_t stream)
{
    const float* DNAs  = (const float*)d_in[0];
    const float* Wpre  = (const float*)d_in[1];
    const float* bpre  = (const float*)d_in[2];
    const float* Wq    = (const float*)d_in[3];
    const float* bq    = (const float*)d_in[4];
    const float* Wk    = (const float*)d_in[5];
    const float* bk    = (const float*)d_in[6];
    const float* Wv    = (const float*)d_in[7];
    const float* bv    = (const float*)d_in[8];
    const float* Wo    = (const float*)d_in[9];
    const float* bo    = (const float*)d_in[10];
    const float* ln1g  = (const float*)d_in[11];
    const float* ln1b  = (const float*)d_in[12];
    const float* W1    = (const float*)d_in[13];
    const float* b1    = (const float*)d_in[14];
    const float* W2    = (const float*)d_in[15];
    const float* b2    = (const float*)d_in[16];
    const float* ln2g  = (const float*)d_in[17];
    const float* ln2b  = (const float*)d_in[18];
    const float* lnfg  = (const float*)d_in[19];
    const float* lnfb  = (const float*)d_in[20];
    const float* Wout  = (const float*)d_in[21];
    const float* bout  = (const float*)d_in[22];
    const int*   sidx  = (const int*)d_in[23];

    const size_t MB = 1024ull * 1024ull;
    char* ws = (char*)d_ws;
    float* x   = (float*)(ws + 0 * MB);    // 16 MB  activations
    float* qb  = (float*)(ws + 16 * MB);   // 16 MB
    float* kb  = (float*)(ws + 32 * MB);   // 16 MB
    float* vb  = (float*)(ws + 48 * MB);   // 16 MB
    float* ctx = (float*)(ws + 64 * MB);   // 16 MB
    float* tmp = (float*)(ws + 80 * MB);   // 16 MB  attn-out / ffn-out
    float* hid = (float*)(ws + 16 * MB);   // 64 MB  aliases q/k/v/ctx (dead by then)
    float* Mb  = (float*)(ws + 96 * MB);   // 1 MB   sparsity measure
    float* vmn = Mb + (size_t)B_ * H_ * L_;
    int*   top = (int*)(vmn + (size_t)B_ * H_ * DK_);

    dim3 blk(256);
    embed_kernel<<<R_ * DM_ / 256, blk, 0, stream>>>(DNAs, Wpre, bpre, x);

    for (int e = 0; e < E_; ++e) {
        const size_t wOff = (size_t)e * DM_ * DM_;
        dim3 g512(R_ / 128, DM_ / 128);     // 64 x 4

        gemm_wmma_bf16<<<g512, blk, 0, stream>>>(x, Wq + wOff, bq + e * DM_, qb, R_, DM_, DM_, 0);
        gemm_wmma_bf16<<<g512, blk, 0, stream>>>(x, Wk + wOff, bk + e * DM_, kb, R_, DM_, DM_, 0);
        gemm_wmma_bf16<<<g512, blk, 0, stream>>>(x, Wv + wOff, bv + e * DM_, vb, R_, DM_, DM_, 0);

        measure_kernel<<<B_ * H_ * L_ / 256, blk, 0, stream>>>(qb, kb, sidx, Mb);
        topk_kernel<<<B_ * H_, blk, 0, stream>>>(Mb, top);
        vmean_kernel<<<B_ * H_ * DK_ / 256, blk, 0, stream>>>(vb, vmn);
        ctxfill_kernel<<<R_ * DM_ / 256, blk, 0, stream>>>(vmn, ctx);
        attn_update_kernel<<<B_ * H_ * U_, blk, 0, stream>>>(qb, kb, vb, top, ctx);

        gemm_wmma_bf16<<<g512, blk, 0, stream>>>(ctx, Wo + wOff, bo + e * DM_, tmp, R_, DM_, DM_, 0);
        add_ln_kernel<<<R_, blk, 0, stream>>>(x, tmp, ln1g + e * DM_, ln1b + e * DM_);

        gemm_wmma_bf16<<<dim3(R_ / 128, DFF_ / 128), blk, 0, stream>>>(
            x, W1 + (size_t)e * DM_ * DFF_, b1 + e * DFF_, hid, R_, DFF_, DM_, 1);
        gemm_wmma_bf16<<<g512, blk, 0, stream>>>(
            hid, W2 + (size_t)e * DFF_ * DM_, b2 + e * DM_, tmp, R_, DM_, DFF_, 0);
        add_ln_kernel<<<R_, blk, 0, stream>>>(x, tmp, ln2g + e * DM_, ln2b + e * DM_);
    }

    final_kernel<<<R_, blk, 0, stream>>>(x, lnfg, lnfb, Wout, bout, (float*)d_out);
}
